// BiMamba_25202868093360
// MI455X (gfx1250) — matrix operand order repaired and checked by reading, not compiled, not run on hardware
//
#include <hip/hip_runtime.h>
#include <hip/hip_bf16.h>
#include <stdint.h>

#define D_MODEL 1024
#define D_INNER 2048
#define D_STATE 16
#define DT_RANK 64
#define SEQ_L   2048
#define XDBL_N  96      // DT_RANK + 2*D_STATE
#define NCHUNK  16
#define LCHUNK  (SEQ_L / NCHUNK)

typedef __bf16 bf16_t;
typedef __attribute__((ext_vector_type(16))) __bf16 v16bf;
typedef __attribute__((ext_vector_type(8)))  __bf16 v8bf;
typedef __attribute__((ext_vector_type(8)))  float  v8f;

// ---------------- helpers ----------------
__device__ __forceinline__ unsigned short f32_to_bf16_rne(float f) {
  unsigned int u = __float_as_uint(f);
  u += 0x7fffu + ((u >> 16) & 1u);
  return (unsigned short)(u >> 16);
}
__device__ __forceinline__ float silu(float v) {
  return v / (1.0f + __expf(-v));
}

// ---------------- fp32 -> bf16 (strided 2D) ----------------
__global__ __launch_bounds__(256)
void k_f32_to_bf16(const float* __restrict__ src, unsigned short* __restrict__ dst,
                   int rows, int cols, int lds, int ldd) {
  int idx = blockIdx.x * blockDim.x + threadIdx.x;
  if (idx >= rows * cols) return;
  int r = idx / cols, c = idx - r * cols;
  dst[(size_t)r * ldd + c] = f32_to_bf16_rne(src[(size_t)r * lds + c]);
}

// ---------------- WMMA bf16 GEMM:  C[M,N] = A[M,K] @ W[N,K]^T ----------------
// All leading dims / K are compile-time: fragment loads and the strided
// epilogue stores become base + immediate-offset memory ops.
// Wave tile: (16*TM) x (16*TN). Block: 8 waves as 4x2 -> (64*TM) x (32*TN).
// MODE 0: store fp32.  MODE 1: softplus(acc + bias[n]).
template <int LD>
__device__ __forceinline__ v16bf load_fragA(const bf16_t* __restrict__ rowp, int k0) {
  // rowp = &A[row * LD + 8*g], precomputed per lane
  v8bf lo = *(const v8bf*)(rowp + k0);        // K = 8g .. 8g+7       -> v0..v3
  v8bf hi = *(const v8bf*)(rowp + k0 + 16);   // K = 16+8g .. 16+8g+7 -> v4..v7
  v16bf a;
#pragma unroll
  for (int i = 0; i < 8; ++i) { a[i] = lo[i]; a[i + 8] = hi[i]; }
  return a;
}
template <int LD>
__device__ __forceinline__ v16bf load_fragB(const bf16_t* __restrict__ colp, int k0) {
  // colp = &W[col * LD + 8*g], precomputed per lane
  v8bf lo = *(const v8bf*)(colp + k0);        // K = 8g .. 8g+7
  v8bf hi = *(const v8bf*)(colp + k0 + 16);   // K = 16+8g .. 16+8g+7
  v16bf b;
#pragma unroll
  for (int i = 0; i < 8; ++i) { b[i] = lo[i]; b[i + 8] = hi[i]; }
  return b;
}

template <int TM, int TN, int MODE, int K, int LDA, int LDW, int LDC>
__global__ __launch_bounds__(256)
void k_gemm(const bf16_t* __restrict__ A, const bf16_t* __restrict__ W,
            float* __restrict__ C, const float* __restrict__ bias) {
  const int lane = threadIdx.x & 31;
  const int wave = threadIdx.x >> 5;
  const int wr = wave & 3, wc = wave >> 2;
  const int m0 = blockIdx.y * (64 * TM) + wr * (16 * TM);
  const int n0 = blockIdx.x * (32 * TN) + wc * (16 * TN);
  const int g = lane >> 4;
  const int l15 = lane & 15;

  // per-lane base pointers (one-time 64-bit math; loop uses immediate offsets)
  const bf16_t* ap[TM];
#pragma unroll
  for (int i = 0; i < TM; ++i)
    ap[i] = A + (size_t)(m0 + 16 * i + l15) * LDA + 8 * g;
  const bf16_t* bp[TN];
#pragma unroll
  for (int j = 0; j < TN; ++j)
    bp[j] = W + (size_t)(n0 + 16 * j + l15) * LDW + 8 * g;

  v8f acc[TM][TN];
#pragma unroll
  for (int i = 0; i < TM; ++i)
#pragma unroll
    for (int j = 0; j < TN; ++j) acc[i][j] = v8f{};

#pragma unroll 2
  for (int k = 0; k < K; k += 32) {
    v16bf a[TM], b[TN];
#pragma unroll
    for (int i = 0; i < TM; ++i) a[i] = load_fragA<LDA>(ap[i], k);
#pragma unroll
    for (int j = 0; j < TN; ++j) b[j] = load_fragB<LDW>(bp[j], k);
#pragma unroll
    for (int i = 0; i < TM; ++i)
#pragma unroll
      for (int j = 0; j < TN; ++j)
        acc[i][j] = __builtin_amdgcn_wmma_f32_16x16x32_bf16(
            false, a[i], false, b[j], (short)0, acc[i][j], false, false);
  }

  // epilogue: C/D f32 layout = 8 VGPRs, lane<16 rows m0..m0+7, lane>=16 rows +8
#pragma unroll
  for (int i = 0; i < TM; ++i) {
#pragma unroll
    for (int j = 0; j < TN; ++j) {
      const int col = n0 + 16 * j + l15;
      float* cp = C + (size_t)(m0 + 16 * i + g * 8) * LDC + col;
      const float bv = (MODE == 1) ? bias[col] : 0.0f;
#pragma unroll
      for (int e = 0; e < 8; ++e) {
        float v = acc[i][j][e];
        if (MODE == 1) {
          v += bv;
          v = (v > 20.0f) ? v : log1pf(__expf(v));   // softplus
        }
        cp[(size_t)e * LDC] = v;                     // immediate offsets (e*LDC*4)
      }
    }
  }
}

// ---------------- causal depthwise conv (D_CONV=4) + SiLU ----------------
__global__ __launch_bounds__(256)
void k_conv_silu(const float* __restrict__ xz, const float* __restrict__ w,
                 const float* __restrict__ b, float* __restrict__ xc,
                 unsigned short* __restrict__ xcbf, int rev) {
  int idx = blockIdx.x * blockDim.x + threadIdx.x;
  if (idx >= SEQ_L * D_INNER) return;
  int l = idx >> 11, d = idx & (D_INNER - 1);
  float acc = b[d];
#pragma unroll
  for (int k = 0; k < 4; ++k) {
    int ls = l - 3 + k;
    if (ls >= 0) {
      int lsrc = rev ? (SEQ_L - 1 - ls) : ls;
      acc += xz[(size_t)lsrc * (2 * D_INNER) + d] * w[d * 4 + k];
    }
  }
  float s = silu(acc);
  xc[idx] = s;
  xcbf[idx] = f32_to_bf16_rne(s);
}

// ---------------- chunked selective scan ----------------
__global__ __launch_bounds__(256)
void k_scan_local(const float* __restrict__ dt, const float* __restrict__ xc,
                  const float* __restrict__ xdbl, const float* __restrict__ A_log,
                  float* __restrict__ Ssum, float* __restrict__ hloc) {
  int idx = blockIdx.x * blockDim.x + threadIdx.x;   // c * D_INNER + d
  if (idx >= NCHUNK * D_INNER) return;
  int c = idx >> 11, d = idx & (D_INNER - 1);
  float A[D_STATE], h[D_STATE];
#pragma unroll
  for (int n = 0; n < D_STATE; ++n) { A[n] = -__expf(A_log[d * D_STATE + n]); h[n] = 0.0f; }
  float S = 0.0f;
  const int l0 = c * LCHUNK;
  for (int t = 0; t < LCHUNK; ++t) {
    int l = l0 + t;
    float dtv = dt[(size_t)l * D_INNER + d];
    float xv  = xc[(size_t)l * D_INNER + d];
    float dx  = dtv * xv;
    S += dtv;
    const float* Bp = xdbl + (size_t)l * XDBL_N + DT_RANK;   // uniform per block -> scalar loads
#pragma unroll
    for (int n = 0; n < D_STATE; ++n) {
      float dA = __expf(dtv * A[n]);
      h[n] = dA * h[n] + dx * Bp[n];
    }
  }
  Ssum[idx] = S;
#pragma unroll
  for (int n = 0; n < D_STATE; ++n) hloc[(size_t)idx * D_STATE + n] = h[n];
}

// h_in(c+1) = exp(A * Ssum(c)) * h_in(c) + hloc(c)
__global__ __launch_bounds__(256)
void k_scan_carry(const float* __restrict__ Ssum, const float* __restrict__ hloc,
                  const float* __restrict__ A_log, float* __restrict__ hin) {
  int d = blockIdx.x * blockDim.x + threadIdx.x;
  if (d >= D_INNER) return;
  float A[D_STATE], h[D_STATE];
#pragma unroll
  for (int n = 0; n < D_STATE; ++n) { A[n] = -__expf(A_log[d * D_STATE + n]); h[n] = 0.0f; }
  for (int c = 0; c < NCHUNK; ++c) {
    size_t base = (size_t)(c * D_INNER + d) * D_STATE;
#pragma unroll
    for (int n = 0; n < D_STATE; ++n) hin[base + n] = h[n];
    float S = Ssum[c * D_INNER + d];
#pragma unroll
    for (int n = 0; n < D_STATE; ++n)
      h[n] = __expf(A[n] * S) * h[n] + hloc[base + n];
  }
}

__global__ __launch_bounds__(256)
void k_scan_final(const float* __restrict__ dt, float* __restrict__ xc,
                  const float* __restrict__ xdbl, const float* __restrict__ A_log,
                  const float* __restrict__ Dvec, const float* __restrict__ hin) {
  int idx = blockIdx.x * blockDim.x + threadIdx.x;
  if (idx >= NCHUNK * D_INNER) return;
  int c = idx >> 11, d = idx & (D_INNER - 1);
  float A[D_STATE], h[D_STATE];
#pragma unroll
  for (int n = 0; n < D_STATE; ++n) A[n] = -__expf(A_log[d * D_STATE + n]);
#pragma unroll
  for (int n = 0; n < D_STATE; ++n) h[n] = hin[(size_t)idx * D_STATE + n];
  float Dd = Dvec[d];
  const int l0 = c * LCHUNK;
  for (int t = 0; t < LCHUNK; ++t) {
    int l = l0 + t;
    float dtv = dt[(size_t)l * D_INNER + d];
    float xv  = xc[(size_t)l * D_INNER + d];
    float dx  = dtv * xv;
    const float* Bp = xdbl + (size_t)l * XDBL_N + DT_RANK;
    const float* Cp = Bp + D_STATE;
    float y = xv * Dd;
#pragma unroll
    for (int n = 0; n < D_STATE; ++n) {
      float dA = __expf(dtv * A[n]);
      h[n] = dA * h[n] + dx * Bp[n];
      y += h[n] * Cp[n];
    }
    xc[(size_t)l * D_INNER + d] = y;   // overwrite x with y (safe: same (l,d))
  }
}

// ---------------- gate + bidirectional combine -> bf16 ----------------
__global__ __launch_bounds__(256)
void k_combine_gate(const float* __restrict__ xz, const float* __restrict__ yf,
                    const float* __restrict__ yr, unsigned short* __restrict__ ycomb) {
  int idx = blockIdx.x * blockDim.x + threadIdx.x;
  if (idx >= SEQ_L * D_INNER) return;
  int l = idx >> 11, d = idx & (D_INNER - 1);
  float z = xz[(size_t)l * (2 * D_INNER) + D_INNER + d];
  float v = silu(z) * (yf[idx] + yr[(size_t)(SEQ_L - 1 - l) * D_INNER + d]);
  ycomb[idx] = f32_to_bf16_rne(v);
}

// ---------------- host orchestration ----------------
static inline void* bump(char*& p, size_t bytes) {
  void* r = (void*)p;
  p += (bytes + 255) & ~(size_t)255;
  return r;
}

extern "C" void kernel_launch(void* const* d_in, const int* in_sizes, int n_in,
                              void* d_out, int out_size, void* d_ws, size_t ws_size,
                              hipStream_t stream) {
  (void)in_sizes; (void)n_in; (void)out_size; (void)ws_size;
  const float* hidden   = (const float*)d_in[0];
  const float* in_w     = (const float*)d_in[1];
  const float* out_w    = (const float*)d_in[2];
  const float* conv_w_f = (const float*)d_in[3];
  const float* conv_b_f = (const float*)d_in[4];
  const float* xp_w_f   = (const float*)d_in[5];
  const float* dt_w_f   = (const float*)d_in[6];
  const float* dt_b_f   = (const float*)d_in[7];
  const float* A_log_f  = (const float*)d_in[8];
  const float* D_f      = (const float*)d_in[9];
  const float* conv_w_r = (const float*)d_in[10];
  const float* conv_b_r = (const float*)d_in[11];
  const float* xp_w_r   = (const float*)d_in[12];
  const float* dt_w_r   = (const float*)d_in[13];
  const float* dt_b_r   = (const float*)d_in[14];
  const float* A_log_r  = (const float*)d_in[15];
  const float* D_r      = (const float*)d_in[16];
  float* out = (float*)d_out;

  char* p = (char*)d_ws;
  unsigned short* hbf    = (unsigned short*)bump(p, (size_t)SEQ_L * D_MODEL * 2);
  unsigned short* wbuf   = (unsigned short*)bump(p, (size_t)(2 * D_INNER) * D_MODEL * 2); // reused for all weights
  float*          xz     = (float*)bump(p, (size_t)SEQ_L * 2 * D_INNER * 4);
  float*          xc_f   = (float*)bump(p, (size_t)SEQ_L * D_INNER * 4);
  float*          xc_r   = (float*)bump(p, (size_t)SEQ_L * D_INNER * 4);
  unsigned short* xcbf   = (unsigned short*)bump(p, (size_t)SEQ_L * D_INNER * 2);         // per-direction reuse
  float*          xdbl_f = (float*)bump(p, (size_t)SEQ_L * XDBL_N * 4);
  float*          xdbl_r = (float*)bump(p, (size_t)SEQ_L * XDBL_N * 4);
  unsigned short* xdblbf = (unsigned short*)bump(p, (size_t)SEQ_L * DT_RANK * 2);         // per-direction reuse
  float*          dt_f   = (float*)bump(p, (size_t)SEQ_L * D_INNER * 4);
  float*          dt_r   = (float*)bump(p, (size_t)SEQ_L * D_INNER * 4);
  float*          Ssum   = (float*)bump(p, (size_t)NCHUNK * D_INNER * 4);
  float*          hloc   = (float*)bump(p, (size_t)NCHUNK * D_INNER * D_STATE * 4);
  float*          hin    = (float*)bump(p, (size_t)NCHUNK * D_INNER * D_STATE * 4);
  unsigned short* ycomb  = (unsigned short*)bump(p, (size_t)SEQ_L * D_INNER * 2);

  const int T = 256;
  auto cvtBlocks = [](size_t n) { return (unsigned)((n + 255) / 256); };

  // 1) bf16 conversions for the big GEMM
  k_f32_to_bf16<<<cvtBlocks((size_t)SEQ_L * D_MODEL), T, 0, stream>>>(hidden, hbf, SEQ_L, D_MODEL, D_MODEL, D_MODEL);
  k_f32_to_bf16<<<cvtBlocks((size_t)2 * D_INNER * D_MODEL), T, 0, stream>>>(in_w, wbuf, 2 * D_INNER, D_MODEL, D_MODEL, D_MODEL);

  // 2) xz = H @ Win^T  (M=2048, N=4096, K=1024), block tile 128x128
  k_gemm<2, 4, 0, D_MODEL, D_MODEL, D_MODEL, 2 * D_INNER>
      <<<dim3((2 * D_INNER) / 128, SEQ_L / 128), T, 0, stream>>>(
      (const bf16_t*)hbf, (const bf16_t*)wbuf, xz, nullptr);

  const unsigned ldBlocks = cvtBlocks((size_t)SEQ_L * D_INNER);
  for (int dir = 0; dir < 2; ++dir) {
    const float* cw  = dir ? conv_w_r : conv_w_f;
    const float* cb  = dir ? conv_b_r : conv_b_f;
    const float* xpw = dir ? xp_w_r  : xp_w_f;
    const float* dtw = dir ? dt_w_r  : dt_w_f;
    const float* dtb = dir ? dt_b_r  : dt_b_f;
    const float* Al  = dir ? A_log_r : A_log_f;
    const float* Dv  = dir ? D_r     : D_f;
    float* xc   = dir ? xc_r   : xc_f;
    float* xdbl = dir ? xdbl_r : xdbl_f;
    float* dt_o = dir ? dt_r   : dt_f;

    // 3) causal depthwise conv + SiLU (fp32 + bf16 copies)
    k_conv_silu<<<ldBlocks, T, 0, stream>>>(xz, cw, cb, xc, xcbf, dir);

    // 4) x_dbl = xc @ x_proj_w^T   (M=2048, N=96, K=2048), block tile 128x96 exact
    k_f32_to_bf16<<<cvtBlocks((size_t)XDBL_N * D_INNER), T, 0, stream>>>(xpw, wbuf, XDBL_N, D_INNER, D_INNER, D_INNER);
    k_gemm<2, 3, 0, D_INNER, D_INNER, D_INNER, XDBL_N>
        <<<dim3(1, SEQ_L / 128), T, 0, stream>>>(
        (const bf16_t*)xcbf, (const bf16_t*)wbuf, xdbl, nullptr);

    // 5) dt = softplus(x_dbl[:, :64] @ dt_proj_w^T + dt_b)  (M=2048, N=2048, K=64)
    k_f32_to_bf16<<<cvtBlocks((size_t)SEQ_L * DT_RANK), T, 0, stream>>>(xdbl, xdblbf, SEQ_L, DT_RANK, XDBL_N, DT_RANK);
    k_f32_to_bf16<<<cvtBlocks((size_t)D_INNER * DT_RANK), T, 0, stream>>>(dtw, wbuf, D_INNER, DT_RANK, DT_RANK, DT_RANK);
    k_gemm<2, 4, 1, DT_RANK, DT_RANK, DT_RANK, D_INNER>
        <<<dim3(D_INNER / 128, SEQ_L / 128), T, 0, stream>>>(
        (const bf16_t*)xdblbf, (const bf16_t*)wbuf, dt_o, dtb);

    // 6) chunked selective scan (y overwrites xc)
    k_scan_local<<<(NCHUNK * D_INNER) / T, T, 0, stream>>>(dt_o, xc, xdbl, Al, Ssum, hloc);
    k_scan_carry<<<D_INNER / T, T, 0, stream>>>(Ssum, hloc, Al, hin);
    k_scan_final<<<(NCHUNK * D_INNER) / T, T, 0, stream>>>(dt_o, xc, xdbl, Al, Dv, hin);
  }

  // 7) gate with silu(z), fold flipped reverse output, convert to bf16
  k_combine_gate<<<ldBlocks, T, 0, stream>>>(xz, xc_f, xc_r, ycomb);

  // 8) out = ycomb @ Wout^T  (M=2048, N=1024, K=2048) -> d_out (fp32)
  k_f32_to_bf16<<<cvtBlocks((size_t)D_MODEL * D_INNER), T, 0, stream>>>(out_w, wbuf, D_MODEL, D_INNER, D_INNER, D_INNER);
  k_gemm<2, 4, 0, D_INNER, D_INNER, D_INNER, D_MODEL>
      <<<dim3(D_MODEL / 128, SEQ_L / 128), T, 0, stream>>>(
      (const bf16_t*)ycomb, (const bf16_t*)wbuf, out, nullptr);
}
